// MixLogisticAttnCoupling_77524159693529
// MI455X (gfx1250) — compile-verified, matrix operand order checked by reading, not run on hardware
//
#include <hip/hip_runtime.h>

// ---------------------------------------------------------------------------
// MixLogisticAttnCoupling forward for gfx1250 (MI455X, wave32, WMMA).
// All matmuls (implicit-GEMM 3x3 convs, nin projections, attention QK^T and
// P*V) run through v_wmma_f32_16x16x32_bf16 with f32 accumulation.
// Weights are pre-packed into WMMA B-fragment order; GEMM waves compute a
// 32x64 output strip (2x4 tiles -> 8 WMMAs per 32-deep k-block) so each
// loadcnt wait is amortized over 8 matrix ops and weight traffic is halved.
// ---------------------------------------------------------------------------

#define NROWS 16384   // B*H*W tokens
#define BATCH 16
#define HH    32
#define WW    32
#define CCH   12
#define FCH   128
#define HEADS 4
#define HDIM  32
#define TSEQ  1024    // H*W
#define OUTCH 1176    // C*(2+3*32)

typedef __attribute__((ext_vector_type(16))) __bf16 v16bf;
typedef __attribute__((ext_vector_type(8)))  float  v8f;

union Frag { v16bf v; unsigned u[8]; unsigned short h[16]; };
union FAcc { v8f   v; float f[8]; };

__device__ __forceinline__ unsigned short f2bf(float f) {
  unsigned x = __float_as_uint(f);
  unsigned r = x + 0x7FFFu + ((x >> 16) & 1u);   // round-to-nearest-even
  return (unsigned short)(r >> 16);
}

// K-index map for 16-bit A/B fragments (ISA 7.12.2, 16-bit 16x32 layout).
__device__ __forceinline__ int kfrag(int v, int half) {
  return ((v & 4) ? 16 : 0) + half * 8 + (v & 3) * 2;
}

// Row-major fragment load: the 8 fragment dwords are two contiguous 16B runs
// at rowbase + half*4 dwords and +8 dwords.  Two b128 loads.
__device__ __forceinline__ void ld_frag(Frag& f, const unsigned* __restrict__ base32,
                                        int half) {
  const uint4* p = (const uint4*)(base32 + half * 4);
  uint4 a = p[0];
  uint4 b = p[2];
  f.u[0] = a.x; f.u[1] = a.y; f.u[2] = a.z; f.u[3] = a.w;
  f.u[4] = b.x; f.u[5] = b.y; f.u[6] = b.z; f.u[7] = b.w;
}

__device__ __forceinline__ void ld_frag_pack(Frag& f, const unsigned* __restrict__ w) {
  const uint4* q = (const uint4*)w;
  uint4 b0 = q[0], b1 = q[1];
  f.u[0] = b0.x; f.u[1] = b0.y; f.u[2] = b0.z; f.u[3] = b0.w;
  f.u[4] = b1.x; f.u[5] = b1.y; f.u[6] = b1.z; f.u[7] = b1.w;
}

__device__ __forceinline__ void zero_frag(Frag& f) {
#pragma unroll
  for (int v = 0; v < 8; v++) f.u[v] = 0u;
}

#define WMMA_BF16(A, B, C) __builtin_amdgcn_wmma_f32_16x16x32_bf16( \
    false, (A), false, (B), (short)0, (C), false, false)

// ---------------------------------------------------------------------------
// Weight pre-pack: f32 [K][F] -> bf16 WMMA B-fragments.
// dst[((kb*T + tn)*32 + lane)*8 + v], T = tiles padded to multiple of 4.
// ---------------------------------------------------------------------------
__global__ void k_pack_w(const float* __restrict__ W, unsigned* __restrict__ dst,
                         int F, int T, long total) {
  long i = (long)blockIdx.x * blockDim.x + threadIdx.x;
  if (i >= total) return;
  int v    = (int)(i & 7);
  int lane = (int)((i >> 3) & 31);
  long rest = i >> 8;
  int tn = (int)(rest % T);
  int kb = (int)(rest / T);
  int half = lane >> 4;
  int col  = tn * 16 + (lane & 15);
  int kk   = kb * 32 + kfrag(v, half);
  unsigned short lo = 0, hi = 0;
  if (col < F) {
    lo = f2bf(W[(long)kk * F + col]);
    hi = f2bf(W[(long)(kk + 1) * F + col]);
  }
  dst[i] = (unsigned)lo | ((unsigned)hi << 16);
}

// ---------------------------------------------------------------------------
// Plain GEMM, K % 32 == 0:  Y[N][F] = A_bf16[N][K] * Wpacked + bias.
// One wave: 32x64 strip (2 row tiles x 4 col tiles), 8 WMMAs per k-block.
// ---------------------------------------------------------------------------
__global__ void k_gemm(const unsigned short* __restrict__ A,
                       const unsigned* __restrict__ Wp,
                       const float* __restrict__ bias,
                       float* __restrict__ Y,
                       int K, int F, int tiles_c4) {
  const int wave = threadIdx.x >> 5;
  const int lane = threadIdx.x & 31;
  const int t    = blockIdx.x * (blockDim.x >> 5) + wave;
  const int tm   = t / tiles_c4;
  if (tm >= (NROWS >> 5)) return;
  const int tn0  = (t - tm * tiles_c4) * 4;
  const int T    = tiles_c4 * 4;
  const int mr   = lane & 15;
  const int half = lane >> 4;

  FAcc acc[2][4];
#pragma unroll
  for (int r = 0; r < 2; r++)
#pragma unroll
    for (int j = 0; j < 4; j++)
#pragma unroll
      for (int v = 0; v < 8; v++) acc[r][j].f[v] = 0.f;

  const unsigned* arow0 = (const unsigned*)A + (long)(tm * 32 + mr) * (K >> 1);
  const unsigned* arow1 = arow0 + (long)16 * (K >> 1);
  const unsigned* wp    = Wp + ((long)tn0 * 32 + lane) * 8;
  const int kblocks = K >> 5;
  for (int kb = 0; kb < kblocks; kb++) {
    Frag fa0, fa1;
    ld_frag(fa0, arow0 + kb * 16, half);
    ld_frag(fa1, arow1 + kb * 16, half);
    const unsigned* w = wp + (long)kb * T * 256;
#pragma unroll
    for (int j = 0; j < 4; j++) {
      Frag fb;
      ld_frag_pack(fb, w + j * 256);
      acc[0][j].v = WMMA_BF16(fa0.v, fb.v, acc[0][j].v);
      acc[1][j].v = WMMA_BF16(fa1.v, fb.v, acc[1][j].v);
    }
  }
#pragma unroll
  for (int r = 0; r < 2; r++) {
#pragma unroll
    for (int j = 0; j < 4; j++) {
      int col = (tn0 + j) * 16 + (lane & 15);
      if (col < F) {
        float bv = bias[col];
#pragma unroll
        for (int v = 0; v < 8; v++) {
          int mm = tm * 32 + r * 16 + v + 8 * half;
          Y[(long)mm * F + col] = acc[r][j].f[v] + bv;
        }
      }
    }
  }
}

// ---------------------------------------------------------------------------
// Implicit-GEMM 3x3 SAME conv, Cin % 32 == 0.  Tap-structured K loop:
// spatial validity is one flag per (tap,row-tile); inner loop pure b128+wmma.
// ---------------------------------------------------------------------------
__global__ void k_conv_gemm(const unsigned short* __restrict__ A,
                            const unsigned* __restrict__ Wp,
                            const float* __restrict__ bias,
                            float* __restrict__ Y,
                            int Cin, int F, int tiles_c4) {
  const int wave = threadIdx.x >> 5;
  const int lane = threadIdx.x & 31;
  const int t    = blockIdx.x * (blockDim.x >> 5) + wave;
  const int tm   = t / tiles_c4;
  if (tm >= (NROWS >> 5)) return;
  const int tn0  = (t - tm * tiles_c4) * 4;
  const int T    = tiles_c4 * 4;
  const int mr   = lane & 15;
  const int half = lane >> 4;

  const int row0 = tm * 32 + mr;
  const int row1 = row0 + 16;
  const int bb0 = row0 >> 10, r0 = row0 & 1023, y0 = r0 >> 5, x0 = r0 & 31;
  const int bb1 = row1 >> 10, r1 = row1 & 1023, y1 = r1 >> 5, x1 = r1 & 31;

  FAcc acc[2][4];
#pragma unroll
  for (int r = 0; r < 2; r++)
#pragma unroll
    for (int j = 0; j < 4; j++)
#pragma unroll
      for (int v = 0; v < 8; v++) acc[r][j].f[v] = 0.f;

  const unsigned* A32 = (const unsigned*)A;
  const unsigned* wp  = Wp + ((long)tn0 * 32 + lane) * 8;
  const int cblocks = Cin >> 5;

#pragma unroll
  for (int s = 0; s < 9; s++) {
    const int dy = s / 3 - 1, dx = s % 3 - 1;
    const int ya = y0 + dy, xa = x0 + dx;
    const int yb = y1 + dy, xb = x1 + dx;
    const bool va = (ya >= 0) && (ya < HH) && (xa >= 0) && (xa < WW);
    const bool vb = (yb >= 0) && (yb < HH) && (xb >= 0) && (xb < WW);
    const unsigned* arow0 = A32 + (((long)((bb0 * HH + ya) * WW + xa) * Cin) >> 1);
    const unsigned* arow1 = A32 + (((long)((bb1 * HH + yb) * WW + xb) * Cin) >> 1);
    for (int cb = 0; cb < cblocks; cb++) {
      Frag fa0, fa1;
      if (va) ld_frag(fa0, arow0 + cb * 16, half); else zero_frag(fa0);
      if (vb) ld_frag(fa1, arow1 + cb * 16, half); else zero_frag(fa1);
      const int kb = s * cblocks + cb;
      const unsigned* w = wp + (long)kb * T * 256;
#pragma unroll
      for (int j = 0; j < 4; j++) {
        Frag fb;
        ld_frag_pack(fb, w + j * 256);
        acc[0][j].v = WMMA_BF16(fa0.v, fb.v, acc[0][j].v);
        acc[1][j].v = WMMA_BF16(fa1.v, fb.v, acc[1][j].v);
      }
    }
  }
#pragma unroll
  for (int r = 0; r < 2; r++) {
#pragma unroll
    for (int j = 0; j < 4; j++) {
      int col = (tn0 + j) * 16 + (lane & 15);
      if (col < F) {
        float bv = bias[col];
#pragma unroll
        for (int v = 0; v < 8; v++) {
          int mm = tm * 32 + r * 16 + v + 8 * half;
          Y[(long)mm * F + col] = acc[r][j].f[v] + bv;
        }
      }
    }
  }
}

// ---------------------------------------------------------------------------
// Generic (slow) GEMM with K guard + per-element conv decode.
// Used ONLY for proj_in (K=108, Cin=12) -- negligible FLOPs.
// ---------------------------------------------------------------------------
__global__ void k_gemm_generic(const unsigned short* __restrict__ A,
                               const unsigned short* __restrict__ Wt,
                               const float* __restrict__ bias,
                               float* __restrict__ Y,
                               int K, int F, int tiles_c, int Cin) {
  const int wave = threadIdx.x >> 5;
  const int lane = threadIdx.x & 31;
  const int tile = blockIdx.x * (blockDim.x >> 5) + wave;
  const int tm = tile / tiles_c;
  if (tm >= (NROWS >> 4)) return;
  const int tn = tile - tm * tiles_c;
  const int mr = lane & 15, half = lane >> 4;
  const int row = tm * 16 + mr;
  const int bb = row >> 10, r = row & 1023, y0 = r >> 5, x0 = r & 31;

  const unsigned* A32 = (const unsigned*)A;
  FAcc acc;
#pragma unroll
  for (int i = 0; i < 8; i++) acc.f[i] = 0.f;

  for (int k0 = 0; k0 < K; k0 += 32) {
    Frag fa, fb;
#pragma unroll
    for (int v = 0; v < 8; v++) {
      int kk = k0 + kfrag(v, half);
      unsigned val = 0u;
      if (kk < K) {
        int s = kk / Cin, cc = kk - s * Cin;
        int yy = y0 + s / 3 - 1, xx = x0 + s % 3 - 1;
        if (yy >= 0 && yy < HH && xx >= 0 && xx < WW)
          val = A32[(((bb * HH + yy) * WW + xx) * Cin + cc) >> 1];
      }
      fa.u[v] = val;
    }
#pragma unroll
    for (int v = 0; v < 8; v++) {
      int kk = k0 + kfrag(v, half);
      int col = tn * 16 + mr;
      unsigned short lo = 0, hi = 0;
      if (kk < K) { lo = Wt[kk * F + col]; hi = Wt[(kk + 1) * F + col]; }
      fb.u[v] = (unsigned)lo | ((unsigned)hi << 16);
    }
    acc.v = WMMA_BF16(fa.v, fb.v, acc.v);
  }
  const int col = tn * 16 + (lane & 15);
  float bv = bias[col];
#pragma unroll
  for (int v = 0; v < 8; v++) {
    int mm = v + 8 * half;
    Y[(long)(tm * 16 + mm) * F + col] = acc.f[v] + bv;
  }
}

// ---------------------------------------------------------------------------
// Flash attention, dim=32, T=1024.  4 waves/block, one 16-query tile per wave.
// q,k: [bh][t][32] bf16; v: TRANSPOSED [bh][d][t] bf16 -> all frags are b128.
// ---------------------------------------------------------------------------
__global__ void k_attn(const unsigned short* __restrict__ qmat,
                       const unsigned short* __restrict__ kmat,
                       const unsigned short* __restrict__ vmat,
                       unsigned short* __restrict__ outp) {
  __shared__ unsigned short Plds[4][16 * 32];
  const int wave = threadIdx.x >> 5;
  const int lane = threadIdx.x & 31;
  const int blk  = blockIdx.x * 4 + wave;   // bh*64 + qtile
  const int bh   = blk >> 6;
  const int qt   = blk & 63;
  const int mr   = lane & 15;
  const int half = lane >> 4;

  const unsigned* q32 = (const unsigned*)qmat;
  const unsigned* k32 = (const unsigned*)kmat;
  const unsigned* v32 = (const unsigned*)vmat;
  const unsigned* p32 = (const unsigned*)&Plds[wave][0];

  Frag fq;
  ld_frag(fq, q32 + ((long)(bh * TSEQ + qt * 16 + mr) * HDIM >> 1), half);

  FAcc O1, O2;
  float M[8], L[8];
#pragma unroll
  for (int v = 0; v < 8; v++) { O1.f[v] = 0.f; O2.f[v] = 0.f; M[v] = -1e30f; L[v] = 0.f; }
  const float scale = 0.17677669529663687f;   // 1/sqrt(32)
  const int n = lane & 15;

  for (int j = 0; j < TSEQ; j += 32) {
    Frag fk1, fk2;
    ld_frag(fk1, k32 + ((long)(bh * TSEQ + j + mr) * HDIM >> 1), half);
    ld_frag(fk2, k32 + ((long)(bh * TSEQ + j + 16 + mr) * HDIM >> 1), half);
    FAcc S1, S2;
#pragma unroll
    for (int v = 0; v < 8; v++) { S1.f[v] = 0.f; S2.f[v] = 0.f; }
    S1.v = WMMA_BF16(fq.v, fk1.v, S1.v);
    S2.v = WMMA_BF16(fq.v, fk2.v, S2.v);

#pragma unroll
    for (int v = 0; v < 8; v++) {
      float s1 = S1.f[v] * scale, s2 = S2.f[v] * scale;
      float t = fmaxf(s1, s2);
      t = fmaxf(t, __shfl_xor(t, 1, 32));
      t = fmaxf(t, __shfl_xor(t, 2, 32));
      t = fmaxf(t, __shfl_xor(t, 4, 32));
      t = fmaxf(t, __shfl_xor(t, 8, 32));
      float nm = fmaxf(M[v], t);
      float em = __expf(M[v] - nm);
      M[v] = nm;
      float p1 = __expf(s1 - nm), p2 = __expf(s2 - nm);
      S1.f[v] = p1; S2.f[v] = p2;
      float rs = p1 + p2;
      rs += __shfl_xor(rs, 1, 32);
      rs += __shfl_xor(rs, 2, 32);
      rs += __shfl_xor(rs, 4, 32);
      rs += __shfl_xor(rs, 8, 32);
      L[v] = L[v] * em + rs;
      O1.f[v] *= em; O2.f[v] *= em;
    }

    __syncthreads();                 // prev-iteration P readers done
#pragma unroll
    for (int v = 0; v < 8; v++) {
      int mm = v + 8 * half;
      Plds[wave][mm * 32 + n]      = f2bf(S1.f[v]);
      Plds[wave][mm * 32 + 16 + n] = f2bf(S2.f[v]);
    }
    __syncthreads();

    Frag fp, fv1, fv2;
    ld_frag(fp, p32 + mr * 16, half);
    // V^T rows: (bh*32 + d), contiguous along keys
    ld_frag(fv1, v32 + (((long)(bh * HDIM + n) * TSEQ + j) >> 1), half);
    ld_frag(fv2, v32 + (((long)(bh * HDIM + 16 + n) * TSEQ + j) >> 1), half);
    O1.v = WMMA_BF16(fp.v, fv1.v, O1.v);
    O2.v = WMMA_BF16(fp.v, fv2.v, O2.v);
  }

  const int b = bh >> 2, head = bh & 3;
#pragma unroll
  for (int v = 0; v < 8; v++) {
    int mm = v + 8 * half;
    float inv = 1.f / L[v];
    int t = qt * 16 + mm;
    outp[(long)(b * TSEQ + t) * FCH + head * HDIM + n]      = f2bf(O1.f[v] * inv);
    outp[(long)(b * TSEQ + t) * FCH + head * HDIM + 16 + n] = f2bf(O2.f[v] * inv);
  }
}

// ---------------------------------------------------------------------------
// Elementwise kernels (f32 VALU)
// ---------------------------------------------------------------------------
__global__ void k_f2bf(const float* __restrict__ s, unsigned short* __restrict__ d, long n) {
  long i = (long)blockIdx.x * blockDim.x + threadIdx.x;
  if (i < n) d[i] = f2bf(s[i]);
}

__global__ void k_concat_elu(const float* __restrict__ in, unsigned short* __restrict__ out,
                             long total) {
  long j = (long)blockIdx.x * blockDim.x + threadIdx.x;
  if (j >= total) return;
  long n = j >> 8;                 // 2F = 256
  int  c = (int)(j & 255);
  float v = (c < FCH) ? in[n * FCH + c] : -in[n * FCH + (c - FCH)];
  float e = (v > 0.f) ? v : (__expf(v) - 1.f);
  out[j] = f2bf(e);
}

__global__ void k_gate_res(float* __restrict__ x, const float* __restrict__ c2, long total) {
  long j = (long)blockIdx.x * blockDim.x + threadIdx.x;
  if (j >= total) return;
  long n = j >> 7;
  int  c = (int)(j & 127);
  float a = c2[n * 256 + c];
  float g = c2[n * 256 + 128 + c];
  x[j] += a / (1.f + __expf(-g));
}

__global__ void k_layernorm(float* __restrict__ x, const float* __restrict__ g,
                            const float* __restrict__ b) {
  int wave = threadIdx.x >> 5, lane = threadIdx.x & 31;
  int row = blockIdx.x * (blockDim.x >> 5) + wave;
  if (row >= NROWS) return;
  float* xr = x + (long)row * FCH;
  float s = 0.f;
  for (int c = lane; c < FCH; c += 32) s += xr[c];
  for (int o = 16; o >= 1; o >>= 1) s += __shfl_xor(s, o, 32);
  float mean = s * (1.f / FCH);
  float vs = 0.f;
  for (int c = lane; c < FCH; c += 32) { float d = xr[c] - mean; vs += d * d; }
  for (int o = 16; o >= 1; o >>= 1) vs += __shfl_xor(vs, o, 32);
  float inv = rsqrtf(vs * (1.f / FCH) + 1e-5f);
  for (int c = lane; c < FCH; c += 32) xr[c] = (xr[c] - mean) * inv * g[c] + b[c];
}

__global__ void k_posadd(const float* __restrict__ x, const float* __restrict__ pos,
                         unsigned short* __restrict__ out, long total) {
  long j = (long)blockIdx.x * blockDim.x + threadIdx.x;
  if (j >= total) return;
  long n = j >> 7;
  int  c = (int)(j & 127);
  out[j] = f2bf(x[j] + pos[(n & 1023) * FCH + c]);
}

__global__ void k_qkv_reorder(const float* __restrict__ qkv,
                              unsigned short* __restrict__ q,
                              unsigned short* __restrict__ k,
                              unsigned short* __restrict__ v, long total) {
  long j = (long)blockIdx.x * blockDim.x + threadIdx.x;
  if (j >= total) return;
  long n  = j / 384;
  int  jj = (int)(j - n * 384);
  int three = jj >> 7, head = (jj >> 5) & 3, d = jj & 31;
  int b = (int)(n >> 10), t = (int)(n & 1023);
  int bh = b * HEADS + head;
  unsigned short val = f2bf(qkv[j]);
  if (three == 0)      q[((long)bh * TSEQ + t) * HDIM + d] = val;
  else if (three == 1) k[((long)bh * TSEQ + t) * HDIM + d] = val;
  else                 v[((long)bh * HDIM + d) * TSEQ + t] = val;   // transposed
}

// ---------------------------------------------------------------------------
// Final mixture-of-logistics + inverse sigmoid + affine; deterministic
// per-batch logp reduction (one block per batch element).
// ---------------------------------------------------------------------------
__global__ void k_mix(const float* __restrict__ yout, const float* __restrict__ ef,
                      float* __restrict__ out, float* __restrict__ logp) {
  __shared__ float red[256];
  const int b = blockIdx.x;
  const int per = HH * WW * CCH;       // 12288
  float acc = 0.f;
  for (int i = threadIdx.x; i < per; i += blockDim.x) {
    long e = (long)b * per + i;
    long n = e / CCH;
    int  c = (int)(e - n * CCH);
    const float* yr = yout + n * OUTCH + c * 98;
    float s  = tanhf(yr[0]);
    float tt = yr[1];
    float lmax = -1e30f;
    for (int q = 0; q < 32; q++) lmax = fmaxf(lmax, yr[2 + q]);
    float lsum = 0.f;
    for (int q = 0; q < 32; q++) lsum += __expf(yr[2 + q] - lmax);
    float lse = __logf(lsum) + lmax;
    float xv = ef[e];
    float mc = -1e30f, sc = 0.f, mp = -1e30f, sp = 0.f;
    for (int q = 0; q < 32; q++) {
      float lw   = yr[2 + q] - lse;
      float mean = yr[34 + q];
      float ls   = fmaxf(yr[66 + q], -7.0f);
      float z    = (xv - mean) * __expf(-ls);
      float spnz = fmaxf(-z, 0.f) + log1pf(__expf(-fabsf(z)));   // softplus(-z)
      float acdf = lw - spnz;
      float apdf = lw - ls - z - 2.f * spnz;
      if (acdf > mc) { sc = sc * __expf(mc - acdf) + 1.f; mc = acdf; }
      else            sc += __expf(acdf - mc);
      if (apdf > mp) { sp = sp * __expf(mp - apdf) + 1.f; mp = apdf; }
      else            sp += __expf(apdf - mp);
    }
    float logcdf = __logf(sc) + mc;
    float logpdf = __logf(sp) + mp;
    float h = __expf(logcdf);
    float y = -__logf(fmaxf(1.f / h - 1.f, 1e-7f));
    out[e] = y * __expf(s) + tt;
    acc += logpdf - __logf(fmaxf(h, 1e-7f)) - __logf(fmaxf(1.f - h, 1e-7f)) + s;
  }
  red[threadIdx.x] = acc;
  __syncthreads();
  for (int o = 128; o > 0; o >>= 1) {
    if ((int)threadIdx.x < o) red[threadIdx.x] += red[threadIdx.x + o];
    __syncthreads();
  }
  if (threadIdx.x == 0) logp[b] = red[0];
}

// ---------------------------------------------------------------------------
// Host orchestration
// ---------------------------------------------------------------------------
static inline char* ws_take(char*& p, size_t bytes) {
  char* r = p;
  p += (bytes + 255) & ~(size_t)255;
  return r;
}
static inline int pad4(int tiles) { return (tiles + 3) & ~3; }

extern "C" void kernel_launch(void* const* d_in, const int* in_sizes, int n_in,
                              void* d_out, int out_size, void* d_ws, size_t ws_size,
                              hipStream_t stream) {
  (void)in_sizes; (void)n_in; (void)out_size; (void)ws_size;
  // setup_inputs() dict order: cf, ef, params{pos_emb, proj_in_W, proj_in_b,
  // proj_out_W, proj_out_b, block0{12}, ..., block3{12}}
  const float* cf    = (const float*)d_in[0];
  const float* ef    = (const float*)d_in[1];
  const float* pos   = (const float*)d_in[2];
  const float* w_in  = (const float*)d_in[3];
  const float* b_in  = (const float*)d_in[4];
  const float* w_out = (const float*)d_in[5];
  const float* b_out = (const float*)d_in[6];

  char* p = (char*)d_ws;
  float* x    = (float*)ws_take(p, (size_t)NROWS * 128 * 4);
  float* h128 = (float*)ws_take(p, (size_t)NROWS * 128 * 4);
  float* hbig = (float*)ws_take(p, (size_t)NROWS * 384 * 4);
  float* yout = (float*)ws_take(p, (size_t)NROWS * OUTCH * 4);
  unsigned short* abf   = (unsigned short*)ws_take(p, (size_t)NROWS * 256 * 2);
  unsigned short* cfbf  = (unsigned short*)ws_take(p, (size_t)NROWS * CCH * 2);
  unsigned short* qbf   = (unsigned short*)ws_take(p, (size_t)BATCH * HEADS * TSEQ * HDIM * 2);
  unsigned short* kbf   = (unsigned short*)ws_take(p, (size_t)BATCH * HEADS * TSEQ * HDIM * 2);
  unsigned short* vbf   = (unsigned short*)ws_take(p, (size_t)BATCH * HEADS * TSEQ * HDIM * 2);
  unsigned short* aobf  = (unsigned short*)ws_take(p, (size_t)NROWS * 128 * 2);
  unsigned short* wbf_in = (unsigned short*)ws_take(p, (size_t)13824 * 2);

  // packed weights: (K/32) * Tpad * 256 dwords each
  auto wp_size = [](int K, int F) { return (size_t)(K / 32) * pad4((F + 15) / 16) * 256 * 4; };
  unsigned* wp_out = (unsigned*)ws_take(p, wp_size(1152, OUTCH));
  unsigned *wp_c1[4], *wp_c2[4], *wp_p1[4], *wp_p2[4];
  for (int i = 0; i < 4; i++) {
    wp_c1[i] = (unsigned*)ws_take(p, wp_size(2304, 128));
    wp_c2[i] = (unsigned*)ws_take(p, wp_size(256, 256));
    wp_p1[i] = (unsigned*)ws_take(p, wp_size(128, 384));
    wp_p2[i] = (unsigned*)ws_take(p, wp_size(128, 256));
  }

  auto cvt = [&](const float* s, unsigned short* d, long n) {
    k_f2bf<<<dim3((unsigned)((n + 255) / 256)), 256, 0, stream>>>(s, d, n);
  };
  auto pack = [&](const float* W, unsigned* dst, int K, int F) {
    int T = pad4((F + 15) / 16);
    long total = (long)(K / 32) * T * 256;
    k_pack_w<<<dim3((unsigned)((total + 255) / 256)), 256, 0, stream>>>(W, dst, F, T, total);
  };
  auto gemm = [&](const unsigned short* A, const unsigned* Wp, const float* bias,
                  float* Y, int K, int F) {
    int tc4 = pad4((F + 15) / 16) / 4;
    long t = (long)(NROWS / 32) * tc4;
    k_gemm<<<dim3((unsigned)((t + 7) / 8)), 256, 0, stream>>>(A, Wp, bias, Y, K, F, tc4);
  };
  auto conv = [&](const unsigned short* A, const unsigned* Wp, const float* bias,
                  float* Y, int Cin, int F) {
    int tc4 = pad4((F + 15) / 16) / 4;
    long t = (long)(NROWS / 32) * tc4;
    k_conv_gemm<<<dim3((unsigned)((t + 7) / 8)), 256, 0, stream>>>(A, Wp, bias, Y, Cin, F, tc4);
  };

  // per-call weight prep (no cached state)
  cvt(w_in, wbf_in, 13824);
  pack(w_out, wp_out, 1152, OUTCH);
  for (int i = 0; i < 4; i++) {
    pack((const float*)d_in[7 + i * 12 + 0], wp_c1[i], 2304, 128);
    pack((const float*)d_in[7 + i * 12 + 2], wp_c2[i], 256, 256);
    pack((const float*)d_in[7 + i * 12 + 6], wp_p1[i], 128, 384);
    pack((const float*)d_in[7 + i * 12 + 8], wp_p2[i], 128, 256);
  }

  // proj_in: 3x3 conv C=12 -> 128 (generic path; K=108 not /32)
  cvt(cf, cfbf, (long)NROWS * CCH);
  {
    long tiles = (long)(NROWS / 16) * 8;
    k_gemm_generic<<<dim3((unsigned)((tiles + 7) / 8)), 256, 0, stream>>>(
        cfbf, wbf_in, b_in, x, 108, 128, 8, CCH);
  }

  const long nF  = (long)NROWS * 128;
  const long n2F = (long)NROWS * 256;
  for (int i = 0; i < 4; i++) {
    const float* c1b  = (const float*)d_in[7 + i * 12 + 1];
    const float* c2b  = (const float*)d_in[7 + i * 12 + 3];
    const float* ln1g = (const float*)d_in[7 + i * 12 + 4];
    const float* ln1b = (const float*)d_in[7 + i * 12 + 5];
    const float* p1b  = (const float*)d_in[7 + i * 12 + 7];
    const float* p2b  = (const float*)d_in[7 + i * 12 + 9];
    const float* ln2g = (const float*)d_in[7 + i * 12 + 10];
    const float* ln2b = (const float*)d_in[7 + i * 12 + 11];

    // gated_conv
    k_concat_elu<<<dim3((unsigned)((n2F + 255) / 256)), 256, 0, stream>>>(x, abf, n2F);
    conv(abf, wp_c1[i], c1b, h128, 256, 128);
    k_concat_elu<<<dim3((unsigned)((n2F + 255) / 256)), 256, 0, stream>>>(h128, abf, n2F);
    gemm(abf, wp_c2[i], c2b, hbig, 256, 256);
    k_gate_res<<<dim3((unsigned)((nF + 255) / 256)), 256, 0, stream>>>(x, hbig, nF);
    k_layernorm<<<dim3(NROWS / 8), 256, 0, stream>>>(x, ln1g, ln1b);

    // gated_attn
    k_posadd<<<dim3((unsigned)((nF + 255) / 256)), 256, 0, stream>>>(x, pos, abf, nF);
    gemm(abf, wp_p1[i], p1b, hbig, 128, 384);
    long nqkv = (long)NROWS * 384;
    k_qkv_reorder<<<dim3((unsigned)((nqkv + 255) / 256)), 256, 0, stream>>>(hbig, qbf, kbf, vbf, nqkv);
    k_attn<<<dim3(BATCH * HEADS * (TSEQ / 16) / 4), 128, 0, stream>>>(qbf, kbf, vbf, aobf);
    gemm(aobf, wp_p2[i], p2b, hbig, 128, 256);
    k_gate_res<<<dim3((unsigned)((nF + 255) / 256)), 256, 0, stream>>>(x, hbig, nF);
    k_layernorm<<<dim3(NROWS / 8), 256, 0, stream>>>(x, ln2g, ln2b);
  }

  // proj_out: 3x3 conv 128 -> 1176
  cvt(x, abf, nF);
  conv(abf, wp_out, b_out, yout, 128, OUTCH);

  // final mixture-of-logistics; out = 196608 floats, logp = 16 floats
  float* outp = (float*)d_out;
  float* logp = outp + (long)BATCH * HH * WW * CCH;
  k_mix<<<dim3(BATCH), 256, 0, stream>>>(yout, ef, outp, logp);
}
